// MiniMHCLM_71416716198106
// MI455X (gfx1250) — matrix operand order repaired and checked by reading, not compiled, not run on hardware
//
#include <hip/hip_runtime.h>
#include <hip/hip_bf16.h>

#define HC 4
#define CDIM 512
#define KDIM 2048            // HC*CDIM
#define MCOEF 24             // HC*HC + 2*HC
#define NTOK 4096            // B*S
#define VOCAB 32000

#if defined(__has_builtin)
#if __has_builtin(__builtin_amdgcn_global_load_async_to_lds_b128) && \
    __has_builtin(__builtin_amdgcn_s_wait_asynccnt)
#define HAVE_ASYNC 1
#endif
#endif
#ifndef HAVE_ASYNC
#define HAVE_ASYNC 0
#endif

typedef __bf16 bf16_t;
typedef bf16_t v16bf __attribute__((ext_vector_type(16)));
typedef float  v8f   __attribute__((ext_vector_type(8)));
typedef int    v4i   __attribute__((ext_vector_type(4)));

// pointer-to-addrspace-qualified-int4 types for the async-to-LDS builtin
typedef __attribute__((address_space(1))) v4i* g4ptr;
typedef __attribute__((address_space(3))) v4i* l4ptr;

union FragBF {
    v16bf v;
    uint4 q[2];
};

__device__ __forceinline__ unsigned short f2bf(float f) {
    unsigned u = __builtin_bit_cast(unsigned, f);
    unsigned r = u + 0x7FFFu + ((u >> 16) & 1u);   // round-to-nearest-even
    return (unsigned short)(r >> 16);
}

// two f32 -> packed bf16x2 in one dword (lo in [15:0], hi in [31:16])
__device__ __forceinline__ unsigned pack2bf(float lo, float hi) {
    unsigned ul = __builtin_bit_cast(unsigned, lo);
    unsigned uh = __builtin_bit_cast(unsigned, hi);
    ul = ul + 0x7FFFu + ((ul >> 16) & 1u);
    uh = uh + 0x7FFFu + ((uh >> 16) & 1u);
    return (ul >> 16) | (uh & 0xFFFF0000u);
}

__device__ __forceinline__ float sigm(float x) {
    return 1.0f / (1.0f + __expf(-x));
}

// 16B copy global->LDS, async (CDNA5 GLOBAL_LOAD_ASYNC_TO_LDS_B128) when available
__device__ __forceinline__ void cp16_async(const void* g, void* l) {
#if HAVE_ASYNC
    __builtin_amdgcn_global_load_async_to_lds_b128(
        (g4ptr)(unsigned long long)g,
        (l4ptr)(unsigned)(unsigned long long)l, 0, 0);
#else
    *(uint4*)l = *(const uint4*)g;
#endif
}

// ---------------------------------------------------------------------------
// Kernel 1: per-token coefficients (RMS, phi-mix, sigmoid, sinkhorn) + x_in
// ---------------------------------------------------------------------------
__global__ __launch_bounds__(256) void mhc_coeff_kernel(
    const int* __restrict__ ids, const float* __restrict__ embed,
    const float* __restrict__ phi, const float* __restrict__ bvec,
    const float* __restrict__ apre, const float* __restrict__ apost,
    const float* __restrict__ ares,
    float* __restrict__ coeff, unsigned short* __restrict__ xin) {
    __shared__ float xs[KDIM];
    __shared__ float red[8][MCOEF + 1];
    __shared__ float cf[MCOEF];

    const int n = blockIdx.x;
    const int t = threadIdx.x;
    const float* xr = embed + (size_t)ids[n] * KDIM;

    for (int i = t; i < KDIM / 4; i += 256)
        ((float4*)xs)[i] = ((const float4*)xr)[i];
    __syncthreads();

    float acc[MCOEF];
#pragma unroll
    for (int j = 0; j < MCOEF; ++j) acc[j] = 0.f;
    float ssq = 0.f;
    for (int kk = t; kk < KDIM; kk += 256) {
        const float xv = xs[kk];
        ssq += xv * xv;
        const float* pr = phi + (size_t)kk * MCOEF;
#pragma unroll
        for (int j = 0; j < MCOEF; ++j) acc[j] += xv * pr[j];
    }
#pragma unroll
    for (int off = 16; off > 0; off >>= 1) {
        ssq += __shfl_down(ssq, off, 32);
#pragma unroll
        for (int j = 0; j < MCOEF; ++j) acc[j] += __shfl_down(acc[j], off, 32);
    }
    const int lane = t & 31, wv = t >> 5;
    if (lane == 0) {
        red[wv][MCOEF] = ssq;
#pragma unroll
        for (int j = 0; j < MCOEF; ++j) red[wv][j] = acc[j];
    }
    __syncthreads();

    if (t == 0) {
        float mix[MCOEF], s2 = 0.f;
#pragma unroll
        for (int j = 0; j < MCOEF; ++j) mix[j] = 0.f;
        for (int w = 0; w < 8; ++w) {
            s2 += red[w][MCOEF];
#pragma unroll
            for (int j = 0; j < MCOEF; ++j) mix[j] += red[w][j];
        }
        const float invr = rsqrtf(s2 * (1.0f / KDIM) + 1e-6f);
#pragma unroll
        for (int j = 0; j < MCOEF; ++j) mix[j] *= invr;

        const float ap = apre[0], ao = apost[0], ar = ares[0];
#pragma unroll
        for (int h = 0; h < HC; ++h)
            cf[h] = sigm(mix[h] * ap + bvec[h]) + 0.01f;
#pragma unroll
        for (int h = 0; h < HC; ++h)
            cf[HC + h] = sigm(mix[HC + h] * ao + bvec[HC + h]) * 2.0f;

        float sm[16];
#pragma unroll
        for (int o = 0; o < 4; ++o) {
            float L[4], mx = -1e30f;
#pragma unroll
            for (int i = 0; i < 4; ++i) {
                L[i] = mix[8 + o * 4 + i] * ar + bvec[8 + o * 4 + i];
                mx = fmaxf(mx, L[i]);
            }
            float s = 0.f, e[4];
#pragma unroll
            for (int i = 0; i < 4; ++i) { e[i] = __expf(L[i] - mx); s += e[i]; }
            const float inv = 1.0f / s;
#pragma unroll
            for (int i = 0; i < 4; ++i) sm[o * 4 + i] = e[i] * inv + 1e-6f;
        }
#pragma unroll
        for (int i = 0; i < 4; ++i) {
            const float s = sm[i] + sm[4 + i] + sm[8 + i] + sm[12 + i];
            const float inv = 1.0f / (s + 1e-6f);
            sm[i] *= inv; sm[4 + i] *= inv; sm[8 + i] *= inv; sm[12 + i] *= inv;
        }
        for (int it = 0; it < 7; ++it) {
#pragma unroll
            for (int o = 0; o < 4; ++o) {
                const float s = sm[o * 4] + sm[o * 4 + 1] + sm[o * 4 + 2] + sm[o * 4 + 3];
                const float inv = 1.0f / (s + 1e-6f);
                sm[o * 4] *= inv; sm[o * 4 + 1] *= inv; sm[o * 4 + 2] *= inv; sm[o * 4 + 3] *= inv;
            }
#pragma unroll
            for (int i = 0; i < 4; ++i) {
                const float s = sm[i] + sm[4 + i] + sm[8 + i] + sm[12 + i];
                const float inv = 1.0f / (s + 1e-6f);
                sm[i] *= inv; sm[4 + i] *= inv; sm[8 + i] *= inv; sm[12 + i] *= inv;
            }
        }
#pragma unroll
        for (int j = 0; j < 16; ++j) cf[8 + j] = sm[j];
    }
    __syncthreads();

    if (t < MCOEF) coeff[(size_t)n * MCOEF + t] = cf[t];

    for (int c = t; c < CDIM; c += 256) {
        const float v = xs[c] * cf[0] + xs[CDIM + c] * cf[1] +
                        xs[2 * CDIM + c] * cf[2] + xs[3 * CDIM + c] * cf[3];
        xin[(size_t)n * CDIM + c] = f2bf(v);
    }
}

// ---------------------------------------------------------------------------
// Kernel 2/4: bf16 WMMA GEMM  Out[M,N] = A[M,K](bf16) * Bsrc[N,K](f32)^T
// 256 threads = 8 waves (2x4), block tile 128x256, wave tile 64x64, BK=32.
// Double-buffered LDS; A tile staged via GLOBAL_LOAD_ASYNC_TO_LDS_B128.
// A tile: 128 rows * 64B @ stride 80B (10240B); B tile: 256 rows * 64B @ 80B (20480B)
// ---------------------------------------------------------------------------
#define TILE_BYTES 30720
__global__ __launch_bounds__(256) void mhc_gemm_bf16_kernel(
    const unsigned short* __restrict__ A, const float* __restrict__ Bsrc,
    float* __restrict__ Out, int M, int N, int K) {
    __shared__ uint4 smem4[2 * TILE_BYTES / 16];
    char* smem = (char*)smem4;

    const int t = threadIdx.x;
    const int lane = t & 31, wave = t >> 5;
    const int wm = wave >> 2, wn = wave & 3;        // 2 x 4 wave grid
    const int r15 = lane & 15, hi = lane >> 4;
    const size_t m0 = (size_t)blockIdx.x * 128;     // M fastest -> W stripe shared in L2
    const size_t n0 = (size_t)blockIdx.y * 256;
    const int nk = K >> 5;

    const v8f zero8 = {0.f, 0.f, 0.f, 0.f, 0.f, 0.f, 0.f, 0.f};
    v8f acc[4][4];
#pragma unroll
    for (int i = 0; i < 4; ++i)
#pragma unroll
        for (int j = 0; j < 4; ++j) acc[i][j] = zero8;

    auto stage = [&](int k0, int sel) {
        char* As = smem + sel * TILE_BYTES;
        char* Bs = As + 10240;
        // A: pure bf16 copy -> async DMA to LDS (2x 16B per thread)
#pragma unroll
        for (int cc = 0; cc < 2; ++cc) {
            const int c = t + cc * 256;
            const int row = c >> 2, q = c & 3;
            cp16_async(A + (m0 + row) * (size_t)K + k0 + q * 8,
                       As + row * 80 + q * 16);
        }
        // B: f32 -> bf16 convert while staging (one 32-elem row per thread)
        const float* src = Bsrc + (n0 + t) * (size_t)K + k0;
#pragma unroll
        for (int q = 0; q < 8; ++q) {
            const float4 f = ((const float4*)src)[q];
            uint2 p;
            p.x = pack2bf(f.x, f.y);
            p.y = pack2bf(f.z, f.w);
            *(uint2*)(Bs + t * 80 + q * 8) = p;
        }
        if (k0 + 32 < K) __builtin_prefetch(src + 32, 0, 0);   // global_prefetch_b8
    };

    stage(0, 0);
    for (int i = 0; i < nk; ++i) {
        const int cur = i & 1;
        if (i + 1 < nk) {
            stage((i + 1) << 5, cur ^ 1);      // overlap next tile with compute
#if HAVE_ASYNC
            __builtin_amdgcn_s_wait_asynccnt(2);   // in-order: tile i's A copies done
#endif
        } else {
#if HAVE_ASYNC
            __builtin_amdgcn_s_wait_asynccnt(0);
#endif
        }
        __syncthreads();

        const char* As = smem + cur * TILE_BYTES;
        const char* Bs = As + 10240;
        FragBF a[4], b[4];
#pragma unroll
        for (int fm = 0; fm < 4; ++fm) {
            const char* p = As + (wm * 64 + fm * 16 + r15) * 80 + hi * 16;
            a[fm].q[0] = *(const uint4*)p;          // K 0..7  / 8..15
            a[fm].q[1] = *(const uint4*)(p + 32);   // K 16..23 / 24..31
        }
#pragma unroll
        for (int fn = 0; fn < 4; ++fn) {
            const char* p = Bs + (wn * 64 + fn * 16 + r15) * 80 + hi * 16;
            b[fn].q[0] = *(const uint4*)p;          // K 0..7  / 8..15
            b[fn].q[1] = *(const uint4*)(p + 32);   // K 16..23 / 24..31
        }
#pragma unroll
        for (int fm = 0; fm < 4; ++fm)
#pragma unroll
            for (int fn = 0; fn < 4; ++fn)
                acc[fm][fn] = __builtin_amdgcn_wmma_f32_16x16x32_bf16(
                    false, a[fm].v, false, b[fn].v, (short)0, acc[fm][fn], false, false);
        __syncthreads();   // protect next-iteration's staging target buffer
    }

    // epilogue: C/D layout -> VGPR r holds row (r + 8*hi), lane%16 = column
#pragma unroll
    for (int fm = 0; fm < 4; ++fm)
#pragma unroll
        for (int fn = 0; fn < 4; ++fn) {
            float* op = Out + (m0 + wm * 64 + fm * 16 + hi * 8) * (size_t)N
                            + (n0 + wn * 64 + fn * 16 + r15);
#pragma unroll
            for (int r = 0; r < 8; ++r) op[(size_t)r * N] = acc[fm][fn][r];
        }
}

// ---------------------------------------------------------------------------
// Kernel 3: x_merge = h_res . x + h_post * f_out  -> bf16
// ---------------------------------------------------------------------------
__global__ __launch_bounds__(256) void mhc_merge_kernel(
    const int* __restrict__ ids, const float* __restrict__ embed,
    const float* __restrict__ coeff, const float* __restrict__ fout,
    unsigned short* __restrict__ xb) {
    __shared__ float cf[MCOEF];
    const int n = blockIdx.x, t = threadIdx.x;
    if (t < MCOEF) cf[t] = coeff[(size_t)n * MCOEF + t];
    __syncthreads();

    const float* xr = embed + (size_t)ids[n] * KDIM;
    const float* fr = fout + (size_t)n * CDIM;
    for (int c = t; c < CDIM; c += 256) {
        const float x0 = xr[c], x1 = xr[CDIM + c], x2 = xr[2 * CDIM + c], x3 = xr[3 * CDIM + c];
        const float fo = fr[c];
#pragma unroll
        for (int o = 0; o < HC; ++o) {
            const float v = cf[8 + o * 4 + 0] * x0 + cf[8 + o * 4 + 1] * x1 +
                            cf[8 + o * 4 + 2] * x2 + cf[8 + o * 4 + 3] * x3 +
                            cf[HC + o] * fo;
            xb[(size_t)n * KDIM + o * CDIM + c] = f2bf(v);
        }
    }
}

// ---------------------------------------------------------------------------
extern "C" void kernel_launch(void* const* d_in, const int* in_sizes, int n_in,
                              void* d_out, int out_size, void* d_ws, size_t ws_size,
                              hipStream_t stream) {
    (void)in_sizes; (void)n_in; (void)out_size; (void)ws_size;
    const int*   ids     = (const int*)d_in[0];
    const float* embed   = (const float*)d_in[1];
    const float* W_inner = (const float*)d_in[2];
    const float* W_head  = (const float*)d_in[3];
    const float* phi     = (const float*)d_in[4];
    const float* bvec    = (const float*)d_in[5];
    const float* a_pre   = (const float*)d_in[6];
    const float* a_post  = (const float*)d_in[7];
    const float* a_res   = (const float*)d_in[8];
    float* out = (float*)d_out;

    char* w = (char*)d_ws;
    float*          coeff = (float*)(w);                      // 4096*24*4  = 0x60000
    unsigned short* xin   = (unsigned short*)(w + 0x60000);   // 4096*512*2 = 0x400000
    float*          fout  = (float*)(w + 0x460000);           // 4096*512*4 = 0x800000
    unsigned short* xb    = (unsigned short*)(w + 0xC60000);  // 4096*2048*2 = 0x1000000

    mhc_coeff_kernel<<<NTOK, 256, 0, stream>>>(ids, embed, phi, bvec,
                                               a_pre, a_post, a_res, coeff, xin);
    mhc_gemm_bf16_kernel<<<dim3(NTOK / 128, CDIM / 256), 256, 0, stream>>>(
        xin, W_inner, fout, NTOK, CDIM, CDIM);
    mhc_merge_kernel<<<NTOK, 256, 0, stream>>>(ids, embed, coeff, fout, xb);
    mhc_gemm_bf16_kernel<<<dim3(NTOK / 128, VOCAB / 256), 256, 0, stream>>>(
        xb, W_head, out, NTOK, VOCAB, KDIM);
}
